// Def_Block_R_12807592477452
// MI455X (gfx1250) — matrix operand order repaired and checked by reading, not compiled, not run on hardware
//
#include <hip/hip_runtime.h>
#include <math.h>

// ---------------------------------------------------------------------------
// MI455X (gfx1250) implementation, round 4.
// Compute-bound (~14.5 GFLOP of K=1152 GEMMs vs <30MB traffic on 23.3TB/s /
// 192MB L2) -> v_wmma_f32_16x16x32_f16 with fp32 accumulation, wave32.
// Round-4 change: explicit 4-phase software pipeline per K-step, separated by
// sched_barrier(0):
//   P1 frag ds_loads (buf p) | P2 next-tile global loads -> regs |
//   P3 4x back-to-back v_wmma (waits only on P1's DS ops) |
//   P4 wait loadcnt + ds_store staged regs -> buf p^1.
// This keeps global-load latency behind the WMMA group (previously the
// scheduler put staging ds_stores before the compute ds_loads, dragging
// global waits into the s_wait_dscnt in front of every WMMA).
// ---------------------------------------------------------------------------

#define BB   4
#define CC   128
#define HH   64
#define WWD  64
#define HWSZ 4096      // H*W
#define KTOT 1152      // 9 * 128, K order = tap*128 + ci  (block s covers 32)
#define NSTEP 36       // KTOT / 32
#define LDA  40        // padded LDS row stride in f16 elems (80B, 16B-aligned)
#define NT   64        // pixels per block (one full (b,h) row)

#if defined(__has_builtin)
#  if __has_builtin(__builtin_amdgcn_sched_barrier)
#    define SCHED_FENCE() __builtin_amdgcn_sched_barrier(0)
#  endif
#endif
#ifndef SCHED_FENCE
#  define SCHED_FENCE()
#endif

typedef __attribute__((ext_vector_type(16))) _Float16 v16h;
typedef __attribute__((ext_vector_type(8)))  _Float16 v8h;
typedef __attribute__((ext_vector_type(8)))  float    v8f;

union HFrag { v16h v; v8h h[2]; };

struct Staged {      // one K-step tile held in registers between P2 and P4
    v8h   a0, a1;    // two 16B weight chunks (this thread's share of A)
    float bv[8];     // this thread's 8 B-tile values (pre f16-convert)
};

// A-fragment: 16x32 f16 (ISA 7.12.2). Lane l: row m = 16*wid + (l&15),
// half g = l>>4; K(e) = 16*(e>>3) + 8*g + (e&7) -> two aligned 16B chunks.
__device__ __forceinline__ HFrag load_Afrag(const _Float16* Als, int wid, int lane) {
    HFrag fa;
    int row = wid * 16 + (lane & 15);
    const _Float16* ap = &Als[row * LDA + (lane >> 4) * 8];
    fa.h[0] = *(const v8h*)(ap);
    fa.h[1] = *(const v8h*)(ap + 16);
    return fa;
}

// B-fragment: 32x16 f16 (KxN). Lane l: col n = t*16+(l&15),
// K(e) = 16*(e>>3) + 8*(l>>4) + (e&7) -> two aligned 16B chunks of the [n][K] tile.
__device__ __forceinline__ HFrag load_Bfrag(const _Float16* Bls, int lane, int t) {
    HFrag fb;
    int nn = t * 16 + (lane & 15);
    const _Float16* bp = &Bls[nn * LDA + (lane >> 4) * 8];
    fb.h[0] = *(const v8h*)(bp);
    fb.h[1] = *(const v8h*)(bp + 16);
    return fb;
}

// P2 helper: this thread's share of the 128x32 weight tile -> registers.
__device__ __forceinline__ void load_Wchunks(const _Float16* __restrict__ wb,
                                             int K0, int tid, Staged& st) {
    int c0 = tid, c1 = tid + 256;            // 512 chunks == 128 rows * 4
    st.a0 = *(const v8h*)(&wb[(c0 >> 2) * KTOT + K0 + (c0 & 3) * 8]);
    st.a1 = *(const v8h*)(&wb[(c1 >> 2) * KTOT + K0 + (c1 & 3) * 8]);
}

// P4 helper: registers -> LDS (A as 16B vectors, B transposed [n][k] as f16).
__device__ __forceinline__ void store_staged(const Staged& st, _Float16* Als,
                                             _Float16* Bls, int tid) {
    int c0 = tid, c1 = tid + 256;
    *(v8h*)(&Als[(c0 >> 2) * LDA + (c0 & 3) * 8]) = st.a0;
    *(v8h*)(&Als[(c1 >> 2) * LDA + (c1 & 3) * 8]) = st.a1;
    int n = tid & 63, krow = tid >> 6;
#pragma unroll
    for (int j = 0; j < 8; ++j)
        Bls[n * LDA + krow + 4 * j] = (_Float16)st.bv[j];
}

// P3: 10 ds_load_b128 (already issued in P1 by caller) feed 4 WMMAs.
__device__ __forceinline__ void wmma4(const HFrag& fa, const HFrag& fb0,
                                      const HFrag& fb1, const HFrag& fb2,
                                      const HFrag& fb3, v8f acc[4]) {
    acc[0] = __builtin_amdgcn_wmma_f32_16x16x32_f16(
        false, fa.v, false, fb0.v, (short)0, acc[0], false, false);
    acc[1] = __builtin_amdgcn_wmma_f32_16x16x32_f16(
        false, fa.v, false, fb1.v, (short)0, acc[1], false, false);
    acc[2] = __builtin_amdgcn_wmma_f32_16x16x32_f16(
        false, fa.v, false, fb2.v, (short)0, acc[2], false, false);
    acc[3] = __builtin_amdgcn_wmma_f32_16x16x32_f16(
        false, fa.v, false, fb3.v, (short)0, acc[3], false, false);
}

// ---------------------------------------------------------------------------
// Weight repack: [co][ci][3][3] fp32 -> [co][tap*128+ci] f16
// ---------------------------------------------------------------------------
__global__ __launch_bounds__(256) void prep_w_kernel(const float* __restrict__ w,
                                                     _Float16* __restrict__ wb) {
    int idx = blockIdx.x * 256 + threadIdx.x;          // 147456 total
    int co = idx / KTOT;
    int r  = idx % KTOT;
    int kk = r >> 7, ci = r & 127;
    wb[co * KTOT + r] = (_Float16)w[(co * CC + ci) * 9 + kk];
}

// ---------------------------------------------------------------------------
// Offset conv: 18 out channels, direct fp32 (~4.5% of FLOPs), wave-uniform
// weight reads go down the scalar path.
// ---------------------------------------------------------------------------
__global__ __launch_bounds__(256) void offset_conv_kernel(
    const float* __restrict__ x, const float* __restrict__ w_off,
    const float* __restrict__ b_off, float* __restrict__ off) {
    int p  = blockIdx.x * 256 + threadIdx.x;   // 0..16383 (B*H*W)
    int b  = p >> 12;
    int hw = p & 4095;
    int h = hw >> 6, w = hw & 63;
    float acc[18];
#pragma unroll
    for (int c = 0; c < 18; ++c) acc[c] = b_off[c];
    for (int kk = 0; kk < 9; ++kk) {
        int hh = h + kk / 3 - 1, ww = w + kk % 3 - 1;
        if (hh < 0 || hh >= HH || ww < 0 || ww >= WWD) continue;
        const float* xb = x + ((size_t)(b * CC) << 12) + hh * WWD + ww;
        for (int ci = 0; ci < CC; ++ci) {
            float xv = xb[(size_t)ci << 12];
#pragma unroll
            for (int c = 0; c < 18; ++c)
                acc[c] = fmaf(xv, w_off[(c * CC + ci) * 9 + kk], acc[c]);
        }
    }
#pragma unroll
    for (int c = 0; c < 18; ++c)
        off[((size_t)(b * 18 + c) << 12) + hw] = acc[c];
}

// ---------------------------------------------------------------------------
// Implicit-GEMM 3x3 conv (pad 1): block = 128 c_out x 64 pixels (one (b,h)
// row); 36 K-steps; double-buffered LDS, one barrier + 4-phase pipeline per
// step. Branch-free boundary masking (clamp + 0/1 multiply).
// ---------------------------------------------------------------------------
__global__ __launch_bounds__(256) void conv3x3_wmma_kernel(
    const float* __restrict__ src, const _Float16* __restrict__ wb,
    const float* __restrict__ bias, float* __restrict__ yraw) {
    __shared__ _Float16 Als[2][CC * LDA];
    __shared__ _Float16 Bls[2][NT * LDA];
    int g = blockIdx.x;                 // 256 blocks
    int b = g >> 6;
    int h = g & 63;
    int tid = threadIdx.x;
    int wid = tid >> 5, lane = tid & 31;
    v8f acc[4] = {{}, {}, {}, {}};

    auto stage_load = [&](int s, Staged& st) {
        int kk = s >> 2, cib = s & 3;
        load_Wchunks(wb, s * 32, tid, st);
        int ky = kk / 3 - 1, kx = kk % 3 - 1;
        int n = tid & 63, krow = tid >> 6;
        int hh = h + ky, wwp = n + kx;
        float msk = ((hh >= 0) & (hh < HH) & (wwp >= 0) & (wwp < WWD)) ? 1.f : 0.f;
        int hc = hh < 0 ? 0 : (hh > HH - 1 ? HH - 1 : hh);
        int wc = wwp < 0 ? 0 : (wwp > WWD - 1 ? WWD - 1 : wwp);
        const float* base =
            src + ((size_t)(b * CC + cib * 32) << 12) + hc * WWD + wc;
#pragma unroll
        for (int j = 0; j < 8; ++j)
            st.bv[j] = base[(size_t)(krow + 4 * j) << 12] * msk;
    };

    {   // prologue: tile 0
        Staged st;
        stage_load(0, st);
        store_staged(st, Als[0], Bls[0], tid);
    }
#pragma unroll 2
    for (int s = 0; s < NSTEP; ++s) {
        __syncthreads();                      // stage(s) visible; buf (s+1)&1 free
        int p = s & 1;
        // P1: fragment ds_loads from buffer p
        HFrag fa  = load_Afrag(Als[p], wid, lane);
        HFrag fb0 = load_Bfrag(Bls[p], lane, 0);
        HFrag fb1 = load_Bfrag(Bls[p], lane, 1);
        HFrag fb2 = load_Bfrag(Bls[p], lane, 2);
        HFrag fb3 = load_Bfrag(Bls[p], lane, 3);
        SCHED_FENCE();
        // P2: issue next tile's global loads into registers (no waits here)
        Staged st;
        if (s + 1 < NSTEP) stage_load(s + 1, st);
        SCHED_FENCE();
        // P3: 4 back-to-back WMMAs (wait only on P1's DS loads)
        wmma4(fa, fb0, fb1, fb2, fb3, acc);
        SCHED_FENCE();
        // P4: global-load waits + LDS stores land behind the WMMA group
        if (s + 1 < NSTEP) store_staged(st, Als[p ^ 1], Bls[p ^ 1], tid);
    }
    int hi = lane >> 4, nl = lane & 15;
#pragma unroll
    for (int r = 0; r < 8; ++r) {              // C-layout: M = r + 8*hi
        int co  = wid * 16 + r + 8 * hi;
        float bv = bias[co];
        size_t o = ((size_t)(b * CC + co) << 12) + h * WWD;
#pragma unroll
        for (int t = 0; t < 4; ++t)
            yraw[o + t * 16 + nl] = acc[t][r] + bv;
    }
}

// ---------------------------------------------------------------------------
// Per-channel biased batch stats over N,H,W -> {mean, rsqrt(var+eps)}
// ---------------------------------------------------------------------------
__global__ __launch_bounds__(256) void bn_stats_kernel(const float* __restrict__ y,
                                                       float* __restrict__ stats) {
    __shared__ float s1[256], s2[256];
    int c = blockIdx.x;
    float a = 0.f, q = 0.f;
    for (int j = threadIdx.x; j < BB * HWSZ; j += 256) {
        int bb = j >> 12, hw = j & 4095;
        float v = y[((size_t)(bb * CC + c) << 12) + hw];
        a += v; q += v * v;
    }
    s1[threadIdx.x] = a; s2[threadIdx.x] = q;
    __syncthreads();
    for (int st = 128; st > 0; st >>= 1) {
        if ((int)threadIdx.x < st) {
            s1[threadIdx.x] += s1[threadIdx.x + st];
            s2[threadIdx.x] += s2[threadIdx.x + st];
        }
        __syncthreads();
    }
    if (threadIdx.x == 0) {
        float m   = s1[0] * (1.0f / 16384.0f);
        float var = s2[0] * (1.0f / 16384.0f) - m * m;
        stats[2 * c]     = m;
        stats[2 * c + 1] = rsqrtf(var + 1e-5f);
    }
}

__global__ __launch_bounds__(256) void bn_relu_kernel(
    const float* __restrict__ yraw, const float* __restrict__ stats,
    const float* __restrict__ gamma, const float* __restrict__ beta,
    float* __restrict__ out) {
    int idx = blockIdx.x * 256 + threadIdx.x;   // 2M elements
    int c = (idx >> 12) & 127;
    float v = (yraw[idx] - stats[2 * c]) * stats[2 * c + 1] * gamma[c] + beta[c];
    out[idx] = fmaxf(v, 0.f);
}

// ---------------------------------------------------------------------------
// Deformable conv GEMM + fused epilogue (bias + BN2 + residual + ReLU).
// Staging thread owns pixel n = tid&63; its 4 clamped bilinear gather
// indices + blend weights (zero-pad folded in) live in registers, refreshed
// once per tap. Same 4-phase pipeline as the conv kernel.
// ---------------------------------------------------------------------------
__global__ __launch_bounds__(256) void deform_wmma_kernel(
    const float* __restrict__ x, const float* __restrict__ off,
    const _Float16* __restrict__ wbd, const float* __restrict__ bd,
    const float* __restrict__ y2raw, const float* __restrict__ stats2,
    const float* __restrict__ g2, const float* __restrict__ beta2,
    float* __restrict__ out) {
    __shared__ _Float16 Als[2][CC * LDA];
    __shared__ _Float16 Bls[2][NT * LDA];
    int g = blockIdx.x;                 // 256 blocks
    int b = g >> 6;
    int h = g & 63;
    int tid = threadIdx.x;
    int wid = tid >> 5, lane = tid & 31;
    v8f acc[4] = {{}, {}, {}, {}};

    int   ci0 = 0, ci1 = 0, ci2 = 0, ci3 = 0;        // per-thread bilinear state
    float cw0 = 0.f, cw1 = 0.f, cw2 = 0.f, cw3 = 0.f;

    auto stage_load = [&](int s, Staged& st) {
        int kk = s >> 2, cib = s & 3;
        load_Wchunks(wbd, s * 32, tid, st);
        int n = tid & 63, krow = tid >> 6;
        if ((s & 3) == 0) {                 // new tap: refresh bilinear coords
            size_t ob = ((size_t)(b * 18 + 2 * kk) << 12) + h * WWD + n;
            float dy = off[ob];
            float dx = off[ob + HWSZ];      // channel 2*kk+1
            float py = (float)(h + kk / 3 - 1) + dy;
            float px = (float)(n + kk % 3 - 1) + dx;
            float fy = floorf(py), fx = floorf(px);
            float ly = py - fy, lx = px - fx;
            int iy0 = (int)fy, ix0 = (int)fx;
            int iy1 = iy0 + 1, ix1 = ix0 + 1;
            float oky0 = (iy0 >= 0) & (iy0 < HH) ? 1.f : 0.f;
            float oky1 = (iy1 >= 0) & (iy1 < HH) ? 1.f : 0.f;
            float okx0 = (ix0 >= 0) & (ix0 < WWD) ? 1.f : 0.f;
            float okx1 = (ix1 >= 0) & (ix1 < WWD) ? 1.f : 0.f;
            int y0c = iy0 < 0 ? 0 : (iy0 > HH - 1 ? HH - 1 : iy0);
            int y1c = iy1 < 0 ? 0 : (iy1 > HH - 1 ? HH - 1 : iy1);
            int x0c = ix0 < 0 ? 0 : (ix0 > WWD - 1 ? WWD - 1 : ix0);
            int x1c = ix1 < 0 ? 0 : (ix1 > WWD - 1 ? WWD - 1 : ix1);
            ci0 = y0c * WWD + x0c;  cw0 = (1.f - ly) * (1.f - lx) * oky0 * okx0;
            ci1 = y0c * WWD + x1c;  cw1 = (1.f - ly) * lx         * oky0 * okx1;
            ci2 = y1c * WWD + x0c;  cw2 = ly         * (1.f - lx) * oky1 * okx0;
            ci3 = y1c * WWD + x1c;  cw3 = ly         * lx         * oky1 * okx1;
        }
        const float* base = x + ((size_t)(b * CC + cib * 32) << 12);
#pragma unroll
        for (int j = 0; j < 8; ++j) {
            const float* bk = base + ((size_t)(krow + 4 * j) << 12);
            st.bv[j] = cw0 * bk[ci0] + cw1 * bk[ci1] + cw2 * bk[ci2] + cw3 * bk[ci3];
        }
    };

    {   // prologue: tile 0
        Staged st;
        stage_load(0, st);
        store_staged(st, Als[0], Bls[0], tid);
    }
#pragma unroll 2
    for (int s = 0; s < NSTEP; ++s) {
        __syncthreads();
        int p = s & 1;
        HFrag fa  = load_Afrag(Als[p], wid, lane);          // P1
        HFrag fb0 = load_Bfrag(Bls[p], lane, 0);
        HFrag fb1 = load_Bfrag(Bls[p], lane, 1);
        HFrag fb2 = load_Bfrag(Bls[p], lane, 2);
        HFrag fb3 = load_Bfrag(Bls[p], lane, 3);
        SCHED_FENCE();
        Staged st;                                          // P2
        if (s + 1 < NSTEP) stage_load(s + 1, st);
        SCHED_FENCE();
        wmma4(fa, fb0, fb1, fb2, fb3, acc);                 // P3
        SCHED_FENCE();
        if (s + 1 < NSTEP) store_staged(st, Als[p ^ 1], Bls[p ^ 1], tid);  // P4
    }
    int hi = lane >> 4, nl = lane & 15;
#pragma unroll
    for (int r = 0; r < 8; ++r) {
        int co = wid * 16 + r + 8 * hi;
        float m = stats2[2 * co], inv = stats2[2 * co + 1];
        float gg = g2[co], bb = beta2[co], bdv = bd[co];
        size_t o = ((size_t)(b * CC + co) << 12) + h * WWD;
#pragma unroll
        for (int t = 0; t < 4; ++t) {
            float d  = acc[t][r] + bdv;
            float o2 = (y2raw[o + t * 16 + nl] - m) * inv * gg + bb;
            out[o + t * 16 + nl] = fmaxf(d + o2, 0.f);
        }
    }
}

// ---------------------------------------------------------------------------
extern "C" void kernel_launch(void* const* d_in, const int* in_sizes, int n_in,
                              void* d_out, int out_size, void* d_ws, size_t ws_size,
                              hipStream_t stream) {
    (void)in_sizes; (void)n_in; (void)out_size; (void)ws_size;
    const float* x     = (const float*)d_in[0];
    const float* w_off = (const float*)d_in[1];
    const float* b_off = (const float*)d_in[2];
    const float* w_d   = (const float*)d_in[3];
    const float* b_d   = (const float*)d_in[4];
    const float* w1    = (const float*)d_in[5];
    const float* b1    = (const float*)d_in[6];
    const float* g1    = (const float*)d_in[7];
    const float* be1   = (const float*)d_in[8];
    const float* w2    = (const float*)d_in[9];
    const float* b2    = (const float*)d_in[10];
    const float* g2    = (const float*)d_in[11];
    const float* be2   = (const float*)d_in[12];
    float* out = (float*)d_out;

    char* ws = (char*)d_ws;
    size_t o = 0;
    const size_t WBYTES = (size_t)CC * KTOT * sizeof(_Float16);   // 294912
    _Float16* wb1 = (_Float16*)(ws + o); o += WBYTES;
    _Float16* wb2 = (_Float16*)(ws + o); o += WBYTES;
    _Float16* wbd = (_Float16*)(ws + o); o += WBYTES;
    float* offb = (float*)(ws + o); o += (size_t)BB * 18 * HWSZ * 4;     // 1.1MB
    float* y1r  = (float*)(ws + o); o += (size_t)BB * CC * HWSZ * 4;     // 8MB
    float* y1n  = (float*)(ws + o); o += (size_t)BB * CC * HWSZ * 4;     // 8MB
    float* y2r  = (float*)(ws + o); o += (size_t)BB * CC * HWSZ * 4;     // 8MB
    float* st1  = (float*)(ws + o); o += 256 * 4;
    float* st2  = (float*)(ws + o); o += 256 * 4;                        // ~27.2MB

    prep_w_kernel<<<576, 256, 0, stream>>>(w1,  wb1);
    prep_w_kernel<<<576, 256, 0, stream>>>(w2,  wb2);
    prep_w_kernel<<<576, 256, 0, stream>>>(w_d, wbd);

    offset_conv_kernel<<<64, 256, 0, stream>>>(x, w_off, b_off, offb);

    conv3x3_wmma_kernel<<<256, 256, 0, stream>>>(x, wb1, b1, y1r);
    bn_stats_kernel<<<128, 256, 0, stream>>>(y1r, st1);
    bn_relu_kernel<<<8192, 256, 0, stream>>>(y1r, st1, g1, be1, y1n);

    conv3x3_wmma_kernel<<<256, 256, 0, stream>>>(y1n, wb2, b2, y2r);
    bn_stats_kernel<<<128, 256, 0, stream>>>(y2r, st2);

    deform_wmma_kernel<<<256, 256, 0, stream>>>(x, offb, wbd, b_d, y2r, st2,
                                                g2, be2, out);
}
